// SimilarityPreserving_61933428408579
// MI455X (gfx1250) — matrix operand order repaired and checked by reading, not compiled, not run on hardware
//
#include <hip/hip_runtime.h>
#include <hip/hip_bf16.h>

// SimilarityPreserving loss, fused: never materializes the 8192x8192 grams.
// loss = sum_i [ Sa_i/Na^2 + Sb_i/Nb^2 - 2*Sab_i/(Na*Nb) ] / b^2
// where a = sim_t rows, b = sim_s rows, computed via f16 WMMA GEMMs over
// row-normalized inputs (1/temperature folded into the X side).

typedef _Float16 half16 __attribute__((ext_vector_type(16)));
typedef _Float16 half8  __attribute__((ext_vector_type(8)));
typedef _Float16 half4  __attribute__((ext_vector_type(4)));
typedef float    v8f    __attribute__((ext_vector_type(8)));

#define BSZ 8192
#define DIM 128

__device__ __forceinline__ half16 cat8(half8 lo, half8 hi) {
  return __builtin_shufflevector(lo, hi, 0, 1, 2, 3, 4, 5, 6, 7,
                                         8, 9, 10, 11, 12, 13, 14, 15);
}

__global__ __launch_bounds__(256) void zero_f32(float* __restrict__ p, int n) {
  int i = blockIdx.x * 256 + threadIdx.x;
  if (i < n) p[i] = 0.0f;
}

// One wave32 per row: load 128 floats (float4/lane), wave-reduce sum of
// squares, write row/||row|| (optionally * 1/temperature) as f16.
__global__ __launch_bounds__(256) void rownorm_f16(const float* __restrict__ src,
                                                   _Float16* __restrict__ dst,
                                                   const float* __restrict__ temp_ptr,
                                                   int fold_temp) {
  const int row  = blockIdx.x * 8 + (threadIdx.x >> 5);
  const int lane = threadIdx.x & 31;
  float4 v = ((const float4*)(src + (size_t)row * DIM))[lane];
  float ss = v.x * v.x + v.y * v.y + v.z * v.z + v.w * v.w;
#pragma unroll
  for (int m = 16; m >= 1; m >>= 1) ss += __shfl_xor(ss, m, 32);
  float scale = 1.0f / sqrtf(ss);            // norms ~ sqrt(128); never 0 here
  if (fold_temp) scale *= 1.0f / temp_ptr[0];
  half4 h;
  h.x = (_Float16)(v.x * scale);
  h.y = (_Float16)(v.y * scale);
  h.z = (_Float16)(v.z * scale);
  h.w = (_Float16)(v.w * scale);
  ((half4*)(dst + (size_t)row * DIM))[lane] = h;
}

// Block: 8 waves, M = 128 rows (wave w owns rows i0+16w..i0+16w+15).
// Each iteration stages a 32-column B slab (Yt & Ys, 32x128 f16 each) in LDS
// shared by all 8 waves, then runs 16 WMMAs across 4 independent accumulator
// chains (ct0/ct1/cs0/cs1). Next slab is register-prefetched during compute.
__global__ __launch_bounds__(256) void gram_kernel(
    const _Float16* __restrict__ Xs, const _Float16* __restrict__ Ys,
    const _Float16* __restrict__ Xt, const _Float16* __restrict__ Yt,
    float* __restrict__ Sa, float* __restrict__ Sb, float* __restrict__ Sab,
    int slabsPerSlice) {
  __shared__ __align__(16) _Float16 sYt[32 * DIM];   // 8 KB
  __shared__ __align__(16) _Float16 sYs[32 * DIM];   // 8 KB

  const int wave = threadIdx.x >> 5;
  const int lane = threadIdx.x & 31;
  const int i0   = blockIdx.x * 128 + wave * 16;

  // A fragments (16x32 f16 per K-chunk). ISA layout: lanes 0-15 hold row M,
  // K in {0..7, 16..23}; lanes 16-31 hold K in {8..15, 24..31}.
  const int arow = i0 + (lane & 15);
  const int klo  = (lane >> 4) * 8;
  half16 At[4], As[4];
#pragma unroll
  for (int kc = 0; kc < 4; ++kc) {
    const _Float16* pt = Xt + (size_t)arow * DIM + kc * 32 + klo;
    At[kc] = cat8(*(const half8*)pt, *(const half8*)(pt + 16));
    const _Float16* ps = Xs + (size_t)arow * DIM + kc * 32 + klo;
    As[kc] = cat8(*(const half8*)ps, *(const half8*)(ps + 16));
  }

  v8f accA = {}, accB = {}, accAB = {};

  const int n   = lane & 15;          // B column within 16-wide tile
  const int kb  = (lane >> 4) * 8;    // B: lanes 0-15 -> K {0..7,16..23}, 16-31 -> K {8..15,24..31}

  // Staging: 32 rows x 128 halves per matrix; thread t moves the contiguous
  // 32 bytes at row (t>>3), half-offset (t&7)*16 (two b128s per matrix).
  const int ldr = threadIdx.x >> 3;          // 0..31
  const int ldc = (threadIdx.x & 7) * 16;    // 0..112

  // Loop-carried global pointers (32 rows per slab), register prefetch.
  const size_t step = (size_t)32 * DIM;
  const _Float16* gYt = Yt + (size_t)(blockIdx.y * slabsPerSlice * 32 + ldr) * DIM + ldc;
  const _Float16* gYs = Ys + (size_t)(blockIdx.y * slabsPerSlice * 32 + ldr) * DIM + ldc;
  half8 rt0 = *(const half8*)gYt;
  half8 rt1 = *(const half8*)(gYt + 8);
  half8 rs0 = *(const half8*)gYs;
  half8 rs1 = *(const half8*)(gYs + 8);

  for (int it = 0; it < slabsPerSlice; ++it) {
    __syncthreads();                           // all waves done with prev slab
    *(half8*)&sYt[ldr * DIM + ldc]     = rt0;  // stage current slab
    *(half8*)&sYt[ldr * DIM + ldc + 8] = rt1;
    *(half8*)&sYs[ldr * DIM + ldc]     = rs0;
    *(half8*)&sYs[ldr * DIM + ldc + 8] = rs1;
    if (it + 1 < slabsPerSlice) {              // prefetch next (overlaps compute)
      gYt += step;
      gYs += step;
      rt0 = *(const half8*)gYt;
      rt1 = *(const half8*)(gYt + 8);
      rs0 = *(const half8*)gYs;
      rs1 = *(const half8*)(gYs + 8);
    }
    __syncthreads();                           // staged slab visible

    v8f ct0 = {}, ct1 = {}, cs0 = {}, cs1 = {};
#pragma unroll
    for (int kc = 0; kc < 4; ++kc) {
      const _Float16* bt0 = &sYt[n * DIM + kc * 32 + kb];
      half16 Bt0 = cat8(*(const half8*)bt0, *(const half8*)(bt0 + 16));
      ct0 = __builtin_amdgcn_wmma_f32_16x16x32_f16(false, At[kc], false, Bt0,
                                                   (short)0, ct0, false, false);
      const _Float16* bt1 = &sYt[(n + 16) * DIM + kc * 32 + kb];
      half16 Bt1 = cat8(*(const half8*)bt1, *(const half8*)(bt1 + 16));
      ct1 = __builtin_amdgcn_wmma_f32_16x16x32_f16(false, At[kc], false, Bt1,
                                                   (short)0, ct1, false, false);
      const _Float16* bs0 = &sYs[n * DIM + kc * 32 + kb];
      half16 Bs0 = cat8(*(const half8*)bs0, *(const half8*)(bs0 + 16));
      cs0 = __builtin_amdgcn_wmma_f32_16x16x32_f16(false, As[kc], false, Bs0,
                                                   (short)0, cs0, false, false);
      const _Float16* bs1 = &sYs[(n + 16) * DIM + kc * 32 + kb];
      half16 Bs1 = cat8(*(const half8*)bs1, *(const half8*)(bs1 + 16));
      cs1 = __builtin_amdgcn_wmma_f32_16x16x32_f16(false, As[kc], false, Bs1,
                                                   (short)0, cs1, false, false);
    }
#pragma unroll
    for (int r = 0; r < 8; ++r) {
      accA[r]  += ct0[r] * ct0[r] + ct1[r] * ct1[r];   // sum_j sim_t^2
      accB[r]  += cs0[r] * cs0[r] + cs1[r] * cs1[r];   // sum_j sim_s^2
      accAB[r] += ct0[r] * cs0[r] + ct1[r] * cs1[r];   // sum_j sim_t*sim_s
    }
  }

  // C layout: VGPR r / lanes 0-15 -> row i0+r, lanes 16-31 -> row i0+8+r.
  // Reduce the 16 N-lanes within each half-wave, then per-row atomics.
#pragma unroll
  for (int r = 0; r < 8; ++r) {
    float sa = accA[r], sb = accB[r], sab = accAB[r];
#pragma unroll
    for (int m = 8; m >= 1; m >>= 1) {
      sa  += __shfl_xor(sa, m, 32);
      sb  += __shfl_xor(sb, m, 32);
      sab += __shfl_xor(sab, m, 32);
    }
    if ((lane & 15) == 0) {
      const int row = i0 + r + ((lane >> 4) ? 8 : 0);
      atomicAdd(&Sa[row], sa);
      atomicAdd(&Sb[row], sb);
      atomicAdd(&Sab[row], sab);
    }
  }
}

__global__ __launch_bounds__(256) void finalize_kernel(const float* __restrict__ Sa,
                                                       const float* __restrict__ Sb,
                                                       const float* __restrict__ Sab,
                                                       float* __restrict__ out) {
  float local = 0.0f;
  for (int i = threadIdx.x; i < BSZ; i += 256) {
    const float sa = Sa[i], sb = Sb[i], sab = Sab[i];
    const float na = fmaxf(sqrtf(sa), 1e-12f);   // torch F.normalize eps
    const float nb = fmaxf(sqrtf(sb), 1e-12f);
    local += sa / (na * na) + sb / (nb * nb) - 2.0f * sab / (na * nb);
  }
  __shared__ float red[256];
  red[threadIdx.x] = local;
  __syncthreads();
  for (int s = 128; s > 0; s >>= 1) {
    if (threadIdx.x < s) red[threadIdx.x] += red[threadIdx.x + s];
    __syncthreads();
  }
  if (threadIdx.x == 0) out[0] = red[0] / ((float)BSZ * (float)BSZ);
}

extern "C" void kernel_launch(void* const* d_in, const int* in_sizes, int n_in,
                              void* d_out, int out_size, void* d_ws, size_t ws_size,
                              hipStream_t stream) {
  const float* zxs  = (const float*)d_in[0];
  const float* zys  = (const float*)d_in[1];
  const float* zxt  = (const float*)d_in[2];
  const float* zyt  = (const float*)d_in[3];
  const float* temp = (const float*)d_in[4];

  char* ws = (char*)d_ws;
  const size_t fbytes = (size_t)BSZ * DIM * sizeof(_Float16);   // 2 MB each
  _Float16* Xs = (_Float16*)(ws + 0 * fbytes);
  _Float16* Ys = (_Float16*)(ws + 1 * fbytes);
  _Float16* Xt = (_Float16*)(ws + 2 * fbytes);
  _Float16* Yt = (_Float16*)(ws + 3 * fbytes);
  float* Sa  = (float*)(ws + 4 * fbytes);
  float* Sb  = Sa + BSZ;
  float* Sab = Sb + BSZ;

  zero_f32<<<(3 * BSZ + 255) / 256, 256, 0, stream>>>(Sa, 3 * BSZ);

  rownorm_f16<<<BSZ / 8, 256, 0, stream>>>(zxs, Xs, temp, 1);
  rownorm_f16<<<BSZ / 8, 256, 0, stream>>>(zys, Ys, temp, 0);
  rownorm_f16<<<BSZ / 8, 256, 0, stream>>>(zxt, Xt, temp, 1);
  rownorm_f16<<<BSZ / 8, 256, 0, stream>>>(zyt, Yt, temp, 0);

  const int nSlices       = 8;                     // j-dim split for occupancy
  const int slabsPerSlice = (BSZ / 32) / nSlices;  // 32 slabs of 32 columns
  dim3 grid(BSZ / 128, nSlices);                   // 64 x 8 = 512 workgroups
  gram_kernel<<<grid, 256, 0, stream>>>(Xs, Ys, Xt, Yt, Sa, Sb, Sab, slabsPerSlice);

  finalize_kernel<<<1, 256, 0, stream>>>(Sa, Sb, Sab, (float*)d_out);
}
